// get_model_21406117004169
// MI455X (gfx1250) — compile-verified, matrix operand order checked by reading, not run on hardware
//
#include <hip/hip_runtime.h>
#include <hip/hip_bf16.h>

// ---------------------------------------------------------------------------
// Types for CDNA5 WMMA
// ---------------------------------------------------------------------------
typedef __attribute__((ext_vector_type(16))) _Float16 v16h;
typedef __attribute__((ext_vector_type(8)))  _Float16 v8h;
typedef __attribute__((ext_vector_type(8)))  float    v8f;

#define TM 64
#define TN 64
#define TK 32

// Load one 16-half WMMA fragment from an LDS row laid out k-contiguous.
// Per the ISA 16-bit operand layout, lane-half hf needs halves
// [hf*8 .. hf*8+7] and [16+hf*8 .. 16+hf*8+7] -> two aligned ds_load_b128.
__device__ __forceinline__ v16h ld_frag(const _Float16* rowp, int hf)
{
    v8h lo = *(const v8h*)(rowp + hf * 8);
    v8h hi = *(const v8h*)(rowp + 16 + hf * 8);
    return __builtin_shufflevector(lo, hi, 0, 1, 2, 3, 4, 5, 6, 7,
                                   8, 9, 10, 11, 12, 13, 14, 15);
}

// ---------------------------------------------------------------------------
// WMMA GEMM:  Y[z] = W[widx] (MxK) * X[xidx] (KxN) + bias[widx]
//   widx = z / wdiv, xidx = z % xmod
// f32 in / f32 out, f16 multiply via v_wmma_f32_16x16x32_f16, f32 accumulate.
// Block = 128 threads = 4 waves; each wave -> 32x32 tile (2x2 WMMA frags).
// Guarded staging is two-phase (all loads issue, then convert/store) so the
// memory pipe stays full even on boundary tiles; interior B tiles use
// float4 (b128/lane) loads.
// ---------------------------------------------------------------------------
__global__ __launch_bounds__(128)
void gemm_wmma(const float* __restrict__ W, const float* __restrict__ Bias,
               const float* __restrict__ X, float* __restrict__ Y,
               int M, int K, int N,
               long wstride, long xstride, long ystride,
               int wdiv, int xmod)
{
    __shared__ _Float16 sA [TM][TK + 8];   // A tile, row-major (m, k)
    __shared__ _Float16 sBT[TN][TK + 8];   // B tile, transposed (n, k)

    const int z    = blockIdx.z;
    const int widx = z / wdiv;
    const int xidx = z % xmod;
    const float* Wp = W + (long)widx * wstride;
    const float* Xp = X + (long)xidx * xstride;
    float*       Yp = Y + (long)z    * ystride;
    const float* bp = Bias ? (Bias + (long)widx * M) : nullptr;

    const int m0   = blockIdx.y * TM;
    const int n0   = blockIdx.x * TN;
    const int tid  = threadIdx.x;
    const int lane = tid & 31;
    const int wave = tid >> 5;
    const int wr   = wave >> 1;   // row half of block tile
    const int wc   = wave & 1;    // col half of block tile
    const int hf   = lane >> 4;   // lane half
    const int mr   = lane & 15;

    const bool mFull = (m0 + TM <= M);
    const bool nFull = (n0 + TN <= N);
    const bool nVec  = nFull && ((N & 3) == 0);

    v8f c00 = {}; v8f c01 = {}; v8f c10 = {}; v8f c11 = {};

    for (int k0 = 0; k0 < K; k0 += TK) {
        const bool kFull = (k0 + TK <= K);

        // ---- stage A tile (64x32)
        if (mFull && kFull) {
#pragma unroll
            for (int it = 0; it < (TM * TK) / 128; ++it) {
                int t = tid + it * 128;
                int r = t >> 5, c = t & 31;
                sA[r][c] = (_Float16)Wp[(long)(m0 + r) * K + (k0 + c)];
            }
        } else {
            // two-phase branchless: issue all clamped loads, then store
            float va[(TM * TK) / 128];
#pragma unroll
            for (int it = 0; it < (TM * TK) / 128; ++it) {
                int t = tid + it * 128;
                int r = t >> 5, c = t & 31;
                int gm = m0 + r, gk = k0 + c;
                bool ok = (gm < M) && (gk < K);
                long idx = (long)(ok ? gm : 0) * K + (ok ? gk : 0);
                va[it] = Wp[idx];
            }
#pragma unroll
            for (int it = 0; it < (TM * TK) / 128; ++it) {
                int t = tid + it * 128;
                int r = t >> 5, c = t & 31;
                int gm = m0 + r, gk = k0 + c;
                bool ok = (gm < M) && (gk < K);
                sA[r][c] = (_Float16)(ok ? va[it] : 0.0f);
            }
        }
        // ---- stage B tile (32x64) transposed into sBT[n][k]
        if (nVec && kFull) {
            // interior: float4 per lane (global_load_b128)
#pragma unroll
            for (int it = 0; it < (TK * TN) / (128 * 4); ++it) {
                int t = tid + it * 128;
                int r  = t >> 4;          // 0..31 (k)
                int c4 = (t & 15) * 4;    // 0..60 (n)
                float4 v = *(const float4*)&Xp[(long)(k0 + r) * N + (n0 + c4)];
                sBT[c4 + 0][r] = (_Float16)v.x;
                sBT[c4 + 1][r] = (_Float16)v.y;
                sBT[c4 + 2][r] = (_Float16)v.z;
                sBT[c4 + 3][r] = (_Float16)v.w;
            }
        } else {
            // two-phase branchless clamped scalar path
            float vb[(TK * TN) / 128];
#pragma unroll
            for (int it = 0; it < (TK * TN) / 128; ++it) {
                int t = tid + it * 128;
                int r = t >> 6, c = t & 63;
                int gk = k0 + r, gn = n0 + c;
                bool ok = (gk < K) && (gn < N);
                long idx = (long)(ok ? gk : 0) * N + (ok ? gn : 0);
                vb[it] = Xp[idx];
            }
#pragma unroll
            for (int it = 0; it < (TK * TN) / 128; ++it) {
                int t = tid + it * 128;
                int r = t >> 6, c = t & 63;
                int gk = k0 + r, gn = n0 + c;
                bool ok = (gk < K) && (gn < N);
                sBT[c][r] = (_Float16)(ok ? vb[it] : 0.0f);
            }
        }
        if (k0 + TK < K) {
            // gfx1250 global_prefetch_b8 on next X tile
            __builtin_prefetch(&Xp[(long)(k0 + TK) * N + n0 + (tid & 63)], 0, 1);
        }
        __syncthreads();

        // ---- fragments: 2x ds_load_b128 each, no repacking
        v16h a0 = ld_frag(&sA [wr * 32 + mr][0],      hf);
        v16h a1 = ld_frag(&sA [wr * 32 + 16 + mr][0], hf);
        v16h b0 = ld_frag(&sBT[wc * 32 + mr][0],      hf);
        v16h b1 = ld_frag(&sBT[wc * 32 + 16 + mr][0], hf);

        c00 = __builtin_amdgcn_wmma_f32_16x16x32_f16(false, a0, false, b0, (short)0, c00, false, false);
        c01 = __builtin_amdgcn_wmma_f32_16x16x32_f16(false, a0, false, b1, (short)0, c01, false, false);
        c10 = __builtin_amdgcn_wmma_f32_16x16x32_f16(false, a1, false, b0, (short)0, c10, false, false);
        c11 = __builtin_amdgcn_wmma_f32_16x16x32_f16(false, a1, false, b1, (short)0, c11, false, false);
        __syncthreads();
    }

    // ---- store (C/D layout: col = lane&15, row = r + 8*(lane>=16))
    const int nn = lane & 15;
#pragma unroll
    for (int r = 0; r < 8; ++r) {
        int lr = hf ? (r + 8) : r;
        int gr, gc;
        gr = m0 + wr * 32 + lr;      gc = n0 + wc * 32 + nn;
        if (gr < M && gc < N) Yp[(long)gr * N + gc] = c00[r] + (bp ? bp[gr] : 0.f);
        gc = n0 + wc * 32 + 16 + nn;
        if (gr < M && gc < N) Yp[(long)gr * N + gc] = c01[r] + (bp ? bp[gr] : 0.f);
        gr = m0 + wr * 32 + 16 + lr; gc = n0 + wc * 32 + nn;
        if (gr < M && gc < N) Yp[(long)gr * N + gc] = c10[r] + (bp ? bp[gr] : 0.f);
        gc = n0 + wc * 32 + 16 + nn;
        if (gr < M && gc < N) Yp[(long)gr * N + gc] = c11[r] + (bp ? bp[gr] : 0.f);
    }
}

// ---------------------------------------------------------------------------
// BatchNorm (training-mode) statistics: one block per (group, channel).
// Y layout: [ngroups*nb][M][Nn]; stats over nb batches x Nn elements.
// ---------------------------------------------------------------------------
__global__ void bn_stats(const float* __restrict__ Y, float* __restrict__ mean,
                         float* __restrict__ var, int M, long Nn, int nb)
{
    int gc = blockIdx.x;
    int g = gc / M, c = gc % M;
    long cstr = (long)M * Nn;
    const float* base = Y + (long)g * nb * cstr + (long)c * Nn;
    float s = 0.f, s2 = 0.f;
    for (int b = 0; b < nb; ++b) {
        const float* p = base + (long)b * cstr;
        for (long i = threadIdx.x; i < Nn; i += blockDim.x) {
            float v = p[i]; s += v; s2 += v * v;
        }
    }
    __shared__ float sh[256], sh2[256];
    int t = threadIdx.x;
    sh[t] = s; sh2[t] = s2;
    __syncthreads();
    for (int d = 128; d > 0; d >>= 1) {
        if (t < d) { sh[t] += sh[t + d]; sh2[t] += sh2[t + d]; }
        __syncthreads();
    }
    if (t == 0) {
        float cnt = (float)((double)nb * (double)Nn);
        float m = sh[0] / cnt;
        mean[gc] = m;
        var[gc] = sh2[0] / cnt - m * m;
    }
}

__global__ void bn_act(float* __restrict__ Y, const float* __restrict__ g,
                       const float* __restrict__ be, const float* __restrict__ mean,
                       const float* __restrict__ var, int M, long Nn, int nb,
                       int relu, long total)
{
    long i = (long)blockIdx.x * blockDim.x + threadIdx.x;
    if (i >= total) return;
    long rest = i / Nn;
    int c = (int)(rest % M);
    long zz = rest / M;
    int grp = (int)(zz / nb);
    int gc = grp * M + c;
    float v = Y[i];
    v = g[gc] * (v - mean[gc]) * rsqrtf(var[gc] + 1e-5f) + be[gc];
    if (relu) v = fmaxf(v, 0.f);
    Y[i] = v;
}

// ---------------------------------------------------------------------------
// Farthest point sampling: one block per batch. xyz: [8][3][n]
// ---------------------------------------------------------------------------
__global__ void fps_kernel(const float* __restrict__ xyz, int* __restrict__ fidx,
                           int n, int npoint)
{
    int b = blockIdx.x;
    const float* X = xyz + (long)b * 3 * n;
    __shared__ float dist[2048];
    __shared__ float rmax[256];
    __shared__ int   rarg[256];
    __shared__ int   sfar;
    int tid = threadIdx.x;
    for (int j = tid; j < n; j += 256) dist[j] = 1e10f;
    if (tid == 0) sfar = 0;
    __syncthreads();
    for (int it = 0; it < npoint; ++it) {
        int far = sfar;
        if (tid == 0) fidx[b * npoint + it] = far;
        float cx = X[far], cy = X[n + far], cz = X[2 * n + far];
        float bm = -1.f; int ba = 0;
        for (int j = tid; j < n; j += 256) {
            float dx = X[j] - cx, dy = X[n + j] - cy, dz = X[2 * n + j] - cz;
            float d = dx * dx + dy * dy + dz * dz;
            float nd = fminf(dist[j], d);
            dist[j] = nd;
            if (nd > bm) { bm = nd; ba = j; }
        }
        rmax[tid] = bm; rarg[tid] = ba;
        __syncthreads();
        for (int s2 = 128; s2 > 0; s2 >>= 1) {
            if (tid < s2 && rmax[tid + s2] > rmax[tid]) {
                rmax[tid] = rmax[tid + s2]; rarg[tid] = rarg[tid + s2];
            }
            __syncthreads();
        }
        if (tid == 0) sfar = rarg[0];
        __syncthreads();
    }
}

__global__ void gather_xyz(const float* __restrict__ xyz, const int* __restrict__ fidx,
                           float* __restrict__ out, int n, int npoint)
{
    int i = blockIdx.x * blockDim.x + threadIdx.x;
    if (i >= 8 * 3 * npoint) return;
    int s = i % npoint;
    int c = (i / npoint) % 3;
    int b = i / (3 * npoint);
    out[i] = xyz[((long)b * 3 + c) * n + fidx[b * npoint + s]];
}

// Ball query: thread per (b, s). First k points within radius, pad with first.
__global__ void ball_query(const float* __restrict__ xyz, const float* __restrict__ newxyz,
                           int* __restrict__ gi, int n, int npoint, int k, float r2)
{
    int i = blockIdx.x * blockDim.x + threadIdx.x;
    if (i >= 8 * npoint) return;
    int b = i / npoint, s = i % npoint;
    const float* X = xyz + (long)b * 3 * n;
    float cx = newxyz[((long)b * 3 + 0) * npoint + s];
    float cy = newxyz[((long)b * 3 + 1) * npoint + s];
    float cz = newxyz[((long)b * 3 + 2) * npoint + s];
    int* g = gi + ((long)b * npoint + s) * k;
    int cnt = 0, first = 0; bool got = false;
    for (int j = 0; j < n && cnt < k; ++j) {
        float dx = X[j] - cx, dy = X[n + j] - cy, dz = X[2 * n + j] - cz;
        if (dx * dx + dy * dy + dz * dz <= r2) {
            if (!got) { first = j; got = true; }
            g[cnt++] = j;
        }
    }
    for (int q = cnt; q < k; ++q) g[q] = first;
}

// Build grouped tensor Xg[b][Cpts+3][k][npoint]
__global__ void group_kernel(const float* __restrict__ pts, const float* __restrict__ xyz,
                             const float* __restrict__ newxyz, const int* __restrict__ gi,
                             float* __restrict__ Xg, int Cpts, int n, int npoint, int k,
                             long total)
{
    long i = (long)blockIdx.x * blockDim.x + threadIdx.x;
    if (i >= total) return;
    int s = (int)(i % npoint);
    long r = i / npoint;
    int kk = (int)(r % k); r /= k;
    int c = (int)(r % (Cpts + 3));
    int b = (int)(r / (Cpts + 3));
    int j = gi[((long)b * npoint + s) * k + kk];
    float v;
    if (c < Cpts) {
        v = pts[((long)b * Cpts + c) * n + j];
    } else {
        int d = c - Cpts;
        v = xyz[((long)b * 3 + d) * n + j] - newxyz[((long)b * 3 + d) * npoint + s];
    }
    Xg[i] = v;
}

// Max over neighborhood dim: src [8][Co][k][npoint] -> dst[b][coff+c][s]
__global__ void max_k(const float* __restrict__ Y, float* __restrict__ dst,
                      int Co, int k, int npoint, int Cdst, int coff, long total)
{
    long i = (long)blockIdx.x * blockDim.x + threadIdx.x;
    if (i >= total) return;
    int s = (int)(i % npoint);
    int c = (int)((i / npoint) % Co);
    int b = (int)(i / ((long)npoint * Co));
    const float* p = Y + ((long)b * Co + c) * (long)k * npoint + s;
    float m = -1e30f;
    for (int kk = 0; kk < k; ++kk) m = fmaxf(m, p[(long)kk * npoint]);
    dst[((long)b * Cdst + coff + c) * npoint + s] = m;
}

// Channel-block copy with optional broadcast over n (srcN==1).
__global__ void copy_ch(const float* __restrict__ src, float* __restrict__ dst,
                        int C, int Nn, int srcN, int Cdst, int coff, long total)
{
    long i = (long)blockIdx.x * blockDim.x + threadIdx.x;
    if (i >= total) return;
    int n = (int)(i % Nn);
    int c = (int)((i / Nn) % C);
    int b = (int)(i / ((long)Nn * C));
    dst[((long)b * Cdst + coff + c) * Nn + n] =
        src[((long)b * C + c) * srcN + (srcN == Nn ? n : 0)];
}

// 3-NN with normalized inverse-distance weights. x1:[8][3][n1], x2:[8][3][s]
__global__ void knn3(const float* __restrict__ x1, const float* __restrict__ x2,
                     int* __restrict__ idx3, float* __restrict__ w3, int n1, int s)
{
    int i = blockIdx.x * blockDim.x + threadIdx.x;
    if (i >= 8 * n1) return;
    int b = i / n1, p = i % n1;
    float px = x1[((long)b * 3 + 0) * n1 + p];
    float py = x1[((long)b * 3 + 1) * n1 + p];
    float pz = x1[((long)b * 3 + 2) * n1 + p];
    float d0 = 1e30f, d1 = 1e30f, d2 = 1e30f;
    int   i0 = 0, i1 = 0, i2 = 0;
    for (int j = 0; j < s; ++j) {
        float dx = x2[((long)b * 3 + 0) * s + j] - px;
        float dy = x2[((long)b * 3 + 1) * s + j] - py;
        float dz = x2[((long)b * 3 + 2) * s + j] - pz;
        float d = dx * dx + dy * dy + dz * dz;
        if (d < d0)      { d2 = d1; i2 = i1; d1 = d0; i1 = i0; d0 = d; i0 = j; }
        else if (d < d1) { d2 = d1; i2 = i1; d1 = d; i1 = j; }
        else if (d < d2) { d2 = d; i2 = j; }
    }
    float w0 = 1.f / (d0 + 1e-8f), w1 = 1.f / (d1 + 1e-8f), w2 = 1.f / (d2 + 1e-8f);
    float wsum = w0 + w1 + w2;
    idx3[(long)i * 3 + 0] = i0; idx3[(long)i * 3 + 1] = i1; idx3[(long)i * 3 + 2] = i2;
    w3[(long)i * 3 + 0] = w0 / wsum; w3[(long)i * 3 + 1] = w1 / wsum; w3[(long)i * 3 + 2] = w2 / wsum;
}

// Weighted gather of p2 [8][C2][s] into dst[b][coff+c][p], p over n1.
__global__ void interp_gather(const float* __restrict__ p2, const int* __restrict__ idx3,
                              const float* __restrict__ w3, float* __restrict__ dst,
                              int C2, int s, int n1, int Cdst, int coff, long total)
{
    long i = (long)blockIdx.x * blockDim.x + threadIdx.x;
    if (i >= total) return;
    int p = (int)(i % n1);
    int c = (int)((i / n1) % C2);
    int b = (int)(i / ((long)n1 * C2));
    const int*   id = idx3 + ((long)b * n1 + p) * 3;
    const float* w  = w3   + ((long)b * n1 + p) * 3;
    const float* F  = p2   + ((long)b * C2 + c) * s;
    dst[((long)b * Cdst + coff + c) * n1 + p] =
        w[0] * F[id[0]] + w[1] * F[id[1]] + w[2] * F[id[2]];
}

// log_softmax over channels + transpose -> out[b][n][C]
__global__ void logsoftmax_t(const float* __restrict__ logits, float* __restrict__ out,
                             int C, int Nn)
{
    int i = blockIdx.x * blockDim.x + threadIdx.x;
    if (i >= 8 * Nn) return;
    int b = i / Nn, n = i % Nn;
    const float* p = logits + (long)b * C * Nn + n;
    float mx = -1e30f;
    for (int c = 0; c < C; ++c) mx = fmaxf(mx, p[(long)c * Nn]);
    float se = 0.f;
    for (int c = 0; c < C; ++c) se += expf(p[(long)c * Nn] - mx);
    float lse = logf(se) + mx;
    float* o = out + ((long)b * Nn + n) * C;
    for (int c = 0; c < C; ++c) o[c] = p[(long)c * Nn] - lse;
}

// mean over n: feat [8][C][Nn] -> z[8*C]; one block per (b,c)
__global__ void mean_n(const float* __restrict__ feat, float* __restrict__ z, long Nn)
{
    long bc = blockIdx.x;
    const float* p = feat + bc * Nn;
    float s = 0.f;
    for (long i = threadIdx.x; i < Nn; i += blockDim.x) s += p[i];
    __shared__ float sh[256];
    int t = threadIdx.x;
    sh[t] = s;
    __syncthreads();
    for (int d = 128; d > 0; d >>= 1) {
        if (t < d) sh[t] += sh[t + d];
        __syncthreads();
    }
    if (t == 0) z[bc] = sh[0] / (float)Nn;
}

// AtlasNet input y[b][2+Cz][g*g]: grid coords then broadcast latent z
__global__ void atlas_y(const float* __restrict__ z, float* __restrict__ y,
                        int Cz, int g, long total)
{
    long i = (long)blockIdx.x * blockDim.x + threadIdx.x;
    if (i >= total) return;
    int npts = g * g;
    int C = Cz + 2;
    int n = (int)(i % npts);
    int c = (int)((i / npts) % C);
    int b = (int)(i / ((long)npts * C));
    float v;
    if (c == 0)      v = (float)(n % g) / (float)(g - 1);
    else if (c == 1) v = (float)(n / g) / (float)(g - 1);
    else             v = z[b * Cz + (c - 2)];
    y[i] = v;
}

__global__ void tanh_k(float* __restrict__ Y, long total)
{
    long i = (long)blockIdx.x * blockDim.x + threadIdx.x;
    if (i >= total) return;
    Y[i] = tanhf(Y[i]);
}

// out[b][p*121+n][c] = Y4[(p*8+b)][c][n]
__global__ void allpts_k(const float* __restrict__ Y4, float* __restrict__ out,
                         int npts, long total)
{
    long i = (long)blockIdx.x * blockDim.x + threadIdx.x;
    if (i >= total) return;
    int c = (int)(i % 3);
    long r = i / 3;
    int n = (int)(r % npts); r /= npts;
    int p = (int)(r % 25);
    int b = (int)(r / 25);
    out[(((long)b * 25 + p) * npts + n) * 3 + c] =
        Y4[(((long)(p * 8 + b)) * 3 + c) * npts + n];
}

// ---------------------------------------------------------------------------
// Host-side orchestration
// ---------------------------------------------------------------------------
struct Layer { const float *w, *b, *g, *be; int ci, co; };

static inline unsigned blocks_for(long total) { return (unsigned)((total + 255) / 256); }

static void conv_bn_act(const Layer& L, const float* X, float* Y, long Nn,
                        int nz, int nb, int ngroups, long wstride, long xstride,
                        int wdiv, int xmod, int relu, float* mv, hipStream_t stream)
{
    dim3 grid((unsigned)((Nn + TN - 1) / TN), (unsigned)((L.co + TM - 1) / TM), (unsigned)nz);
    gemm_wmma<<<grid, 128, 0, stream>>>(L.w, L.b, X, Y, L.co, L.ci, (int)Nn,
                                        wstride, xstride, (long)L.co * Nn, wdiv, xmod);
    if (L.g) {
        bn_stats<<<(unsigned)(ngroups * L.co), 256, 0, stream>>>(Y, mv, mv + 4096, L.co, Nn, nb);
        long total = (long)nz * L.co * Nn;
        bn_act<<<blocks_for(total), 256, 0, stream>>>(Y, L.g, L.be, mv, mv + 4096,
                                                      L.co, Nn, nb, relu, total);
    }
}

static void run_sa_branch(const float* pxyz, const float* pts, int Cpts, int n,
                          const float* newxyz, int npoint, float r, int k,
                          const Layer* layers, int nlayers,
                          float* Xg, float* bufA, float* bufB, int* gi, float* mv,
                          float* dst, int Cdst, int coff, hipStream_t stream)
{
    ball_query<<<blocks_for(8 * npoint), 256, 0, stream>>>(pxyz, newxyz, gi, n, npoint, k, r * r);
    long Nn = (long)k * npoint;
    long totg = 8L * (Cpts + 3) * Nn;
    group_kernel<<<blocks_for(totg), 256, 0, stream>>>(pts, pxyz, newxyz, gi, Xg,
                                                       Cpts, n, npoint, k, totg);
    const float* X = Xg;
    float* Y = bufA;
    for (int li = 0; li < nlayers; ++li) {
        conv_bn_act(layers[li], X, Y, Nn, 8, 8, 1, 0, (long)layers[li].ci * Nn,
                    8, 8, 1, mv, stream);
        X = Y;
        Y = (Y == bufA) ? bufB : bufA;
    }
    long totm = 8L * layers[nlayers - 1].co * npoint;
    max_k<<<blocks_for(totm), 256, 0, stream>>>(X, dst, layers[nlayers - 1].co,
                                                k, npoint, Cdst, coff, totm);
}

extern "C" void kernel_launch(void* const* d_in, const int* in_sizes, int n_in,
                              void* d_out, int out_size, void* d_ws, size_t ws_size,
                              hipStream_t stream)
{
    (void)in_sizes; (void)n_in; (void)out_size; (void)ws_size;

    const float* xyz  = (const float*)d_in[0];   // [8][3][2048]
    const float* clsl = (const float*)d_in[1];   // [8][16]

    int pi = 2;
    auto nxt = [&]() { return (const float*)d_in[pi++]; };
    auto mk  = [&](int ci, int co) {
        Layer l; l.w = nxt(); l.b = nxt(); l.g = nxt(); l.be = nxt();
        l.ci = ci; l.co = co; return l;
    };

    Layer sa1b0[3] = { mk(6, 32),    mk(32, 32),   mk(32, 64)   };
    Layer sa1b1[3] = { mk(6, 64),    mk(64, 64),   mk(64, 128)  };
    Layer sa1b2[3] = { mk(6, 64),    mk(64, 96),   mk(96, 128)  };
    Layer sa2b0[3] = { mk(323, 128), mk(128, 128), mk(128, 256) };
    Layer sa2b1[3] = { mk(323, 128), mk(128, 196), mk(196, 256) };
    Layer sa3l[3]  = { mk(515, 256), mk(256, 512), mk(512, 1024) };
    Layer fp3l[2]  = { mk(1536, 256), mk(256, 256) };
    Layer fp2l[2]  = { mk(576, 256),  mk(256, 128) };
    Layer fp1l[2]  = { mk(150, 128),  mk(128, 128) };
    Layer conv1;  conv1.w = nxt(); conv1.b = nxt();
    conv1.g = nxt(); conv1.be = nxt(); conv1.ci = 128; conv1.co = 128;
    Layer conv2;  conv2.w = nxt(); conv2.b = nxt();
    conv2.g = nullptr; conv2.be = nullptr; conv2.ci = 128; conv2.co = 50;
    const float* dw1 = nxt(); const float* db1 = nxt();
    const float* dg1 = nxt(); const float* de1 = nxt();
    const float* dw2 = nxt(); const float* db2 = nxt();
    const float* dg2 = nxt(); const float* de2 = nxt();
    const float* dw3 = nxt(); const float* db3 = nxt();
    const float* dg3 = nxt(); const float* de3 = nxt();
    const float* dw4 = nxt(); const float* db4 = nxt();

    // ---- workspace bump allocator (floats)
    float* ws = (float*)d_ws;
    size_t off = 0;
    auto alloc = [&](size_t nf) { float* p = ws + off; off += (nf + 63) & ~(size_t)63; return p; };

    int*   fidx1  = (int*)alloc(4096);
    int*   fidx2  = (int*)alloc(1024);
    int*   gi     = (int*)alloc(524288);
    int*   idx3   = (int*)alloc(49152);
    float* w3     = alloc(49152);
    float* mv     = alloc(8192);
    float* nx1    = alloc(12288);           // l1_xyz  [8][3][512]
    float* nx2    = alloc(3072);            // l2_xyz  [8][3][128]
    float* l1p    = alloc(1310720);         // [8][320][512]
    float* l2p    = alloc(524288);          // [8][512][128]
    float* l3p    = alloc(8192);            // [8][1024][1]
    float* fp3o   = alloc(262144);          // [8][256][128]
    float* fp2o   = alloc(524288);          // [8][128][512]
    float* fp1o   = alloc(2097152);         // [8][128][2048]
    float* feat   = alloc(2097152);         // [8][128][2048]
    float* zbuf   = alloc(1024);            // [8][128]
    float* logits = alloc(819200);          // [8][50][2048]
    float* XG     = alloc(42336256);        // concat / grouped input scratch
    float* bufA   = alloc(67108864);
    float* bufB   = alloc(67108864);

    float* out = (float*)d_out;
    const long o_x = 0, o_l1 = 819200, o_l2 = 1343488, o_l3 = 1605632,
               o_feat = 1613824, o_pts = 3710976;

    // =================== SA1 ===================
    fps_kernel<<<8, 256, 0, stream>>>(xyz, fidx1, 2048, 512);
    gather_xyz<<<blocks_for(8 * 3 * 512), 256, 0, stream>>>(xyz, fidx1, nx1, 2048, 512);
    run_sa_branch(xyz, xyz, 3, 2048, nx1, 512, 0.1f, 32,  sa1b0, 3, XG, bufA, bufB, gi, mv, l1p, 320, 0,   stream);
    run_sa_branch(xyz, xyz, 3, 2048, nx1, 512, 0.2f, 64,  sa1b1, 3, XG, bufA, bufB, gi, mv, l1p, 320, 64,  stream);
    run_sa_branch(xyz, xyz, 3, 2048, nx1, 512, 0.4f, 128, sa1b2, 3, XG, bufA, bufB, gi, mv, l1p, 320, 192, stream);

    // =================== SA2 ===================
    fps_kernel<<<8, 256, 0, stream>>>(nx1, fidx2, 512, 128);
    gather_xyz<<<blocks_for(8 * 3 * 128), 256, 0, stream>>>(nx1, fidx2, nx2, 512, 128);
    run_sa_branch(nx1, l1p, 320, 512, nx2, 128, 0.4f, 64,  sa2b0, 3, XG, bufA, bufB, gi, mv, l2p, 512, 0,   stream);
    run_sa_branch(nx1, l1p, 320, 512, nx2, 128, 0.8f, 128, sa2b1, 3, XG, bufA, bufB, gi, mv, l2p, 512, 256, stream);

    // =================== SA3 (global) ===================
    copy_ch<<<blocks_for(8L * 3 * 128), 256, 0, stream>>>(nx2, XG, 3, 128, 128, 515, 0, 8L * 3 * 128);
    copy_ch<<<blocks_for(8L * 512 * 128), 256, 0, stream>>>(l2p, XG, 512, 128, 128, 515, 3, 8L * 512 * 128);
    {
        const float* X = XG; float* Y = bufA;
        for (int li = 0; li < 3; ++li) {
            conv_bn_act(sa3l[li], X, Y, 128, 8, 8, 1, 0, (long)sa3l[li].ci * 128, 8, 8, 1, mv, stream);
            X = Y; Y = (Y == bufA) ? bufB : bufA;
        }
        long totm = 8L * 1024;
        max_k<<<blocks_for(totm), 256, 0, stream>>>(X, l3p, 1024, 128, 1, 1024, 0, totm);
    }

    // =================== FP3 (s==1 broadcast) ===================
    copy_ch<<<blocks_for(8L * 512 * 128), 256, 0, stream>>>(l2p, XG, 512, 128, 128, 1536, 0, 8L * 512 * 128);
    copy_ch<<<blocks_for(8L * 1024 * 128), 256, 0, stream>>>(l3p, XG, 1024, 128, 1, 1536, 512, 8L * 1024 * 128);
    conv_bn_act(fp3l[0], XG,   bufA, 128, 8, 8, 1, 0, 1536L * 128, 8, 8, 1, mv, stream);
    conv_bn_act(fp3l[1], bufA, fp3o, 128, 8, 8, 1, 0, 256L * 128,  8, 8, 1, mv, stream);

    // =================== FP2 ===================
    knn3<<<blocks_for(8 * 512), 256, 0, stream>>>(nx1, nx2, idx3, w3, 512, 128);
    copy_ch<<<blocks_for(8L * 320 * 512), 256, 0, stream>>>(l1p, XG, 320, 512, 512, 576, 0, 8L * 320 * 512);
    interp_gather<<<blocks_for(8L * 256 * 512), 256, 0, stream>>>(fp3o, idx3, w3, XG, 256, 128, 512, 576, 320, 8L * 256 * 512);
    conv_bn_act(fp2l[0], XG,   bufA, 512, 8, 8, 1, 0, 576L * 512, 8, 8, 1, mv, stream);
    conv_bn_act(fp2l[1], bufA, fp2o, 512, 8, 8, 1, 0, 256L * 512, 8, 8, 1, mv, stream);

    // =================== FP1 ===================
    knn3<<<blocks_for(8 * 2048), 256, 0, stream>>>(xyz, nx1, idx3, w3, 2048, 512);
    copy_ch<<<blocks_for(8L * 16 * 2048), 256, 0, stream>>>(clsl, XG, 16, 2048, 1, 150, 0, 8L * 16 * 2048);
    copy_ch<<<blocks_for(8L * 3 * 2048), 256, 0, stream>>>(xyz, XG, 3, 2048, 2048, 150, 16, 8L * 3 * 2048);
    copy_ch<<<blocks_for(8L * 3 * 2048), 256, 0, stream>>>(xyz, XG, 3, 2048, 2048, 150, 19, 8L * 3 * 2048);
    interp_gather<<<blocks_for(8L * 128 * 2048), 256, 0, stream>>>(fp2o, idx3, w3, XG, 128, 512, 2048, 150, 22, 8L * 128 * 2048);
    conv_bn_act(fp1l[0], XG,   bufA, 2048, 8, 8, 1, 0, 150L * 2048, 8, 8, 1, mv, stream);
    conv_bn_act(fp1l[1], bufA, fp1o, 2048, 8, 8, 1, 0, 128L * 2048, 8, 8, 1, mv, stream);

    // =================== head ===================
    conv_bn_act(conv1, fp1o, feat, 2048, 8, 8, 1, 0, 128L * 2048, 8, 8, 1, mv, stream);
    conv_bn_act(conv2, feat, logits, 2048, 8, 8, 1, 0, 128L * 2048, 8, 8, 0, mv, stream);
    logsoftmax_t<<<blocks_for(8 * 2048), 256, 0, stream>>>(logits, out + o_x, 50, 2048);
    mean_n<<<8 * 128, 256, 0, stream>>>(feat, zbuf, 2048);

    // =================== AtlasNet decoder ===================
    atlas_y<<<blocks_for(8L * 130 * 121), 256, 0, stream>>>(zbuf, XG, 128, 11, 8L * 130 * 121);
    const int BIG = 1 << 29;
    {
        Layer a1 = { dw1, db1, dg1, de1, 130, 130 };
        Layer a2 = { dw2, db2, dg2, de2, 130, 65  };
        Layer a3 = { dw3, db3, dg3, de3, 65,  32  };
        Layer a4 = { dw4, db4, nullptr, nullptr, 32, 3 };
        // L1: y broadcast across primitives (xidx = z%8), per-prim weights (widx = z/8)
        conv_bn_act(a1, XG,   bufA, 121, 200, 8, 25, 130L * 130, 130L * 121, 8, 8,   1, mv, stream);
        conv_bn_act(a2, bufA, bufB, 121, 200, 8, 25, 65L * 130,  130L * 121, 8, BIG, 1, mv, stream);
        conv_bn_act(a3, bufB, bufA, 121, 200, 8, 25, 32L * 65,   65L * 121,  8, BIG, 1, mv, stream);
        conv_bn_act(a4, bufA, bufB, 121, 200, 8, 25, 3L * 32,    32L * 121,  8, BIG, 0, mv, stream);
        tanh_k<<<blocks_for(200L * 3 * 121), 256, 0, stream>>>(bufB, 200L * 3 * 121);
        allpts_k<<<blocks_for(8L * 25 * 121 * 3), 256, 0, stream>>>(bufB, out + o_pts, 121, 8L * 25 * 121 * 3);
    }

    // =================== auxiliary outputs ===================
    hipMemcpyAsync(out + o_l1, fp2o, sizeof(float) * 524288,  hipMemcpyDeviceToDevice, stream);
    hipMemcpyAsync(out + o_l2, fp3o, sizeof(float) * 262144,  hipMemcpyDeviceToDevice, stream);
    hipMemcpyAsync(out + o_l3, l3p,  sizeof(float) * 8192,    hipMemcpyDeviceToDevice, stream);
    hipMemcpyAsync(out + o_feat, feat, sizeof(float) * 2097152, hipMemcpyDeviceToDevice, stream);
}
